// completeNet_7026566496517
// MI455X (gfx1250) — matrix operand order repaired and checked by reading, not compiled, not run on hardware
//
#include <hip/hip_runtime.h>
#include <hip/hip_bf16.h>

#define T_NUM   1024
#define DET_NUM 1024
#define DEG     64
#define E_EDGES (T_NUM * DEG)     // 65536 tracklet->det edges
#define NE      (2 * E_EDGES)     // 131072 directed edges
#define NN      (T_NUM + DET_NUM) // 2048 nodes
#define D_IN    512
#define D_EMB   256
#define MROWS   (T_NUM + 1)       // 1025
#define MCOLS   (DET_NUM + 1)     // 1025

typedef _Float16 v16h  __attribute__((ext_vector_type(16)));
typedef _Float16 half8 __attribute__((ext_vector_type(8)));
typedef float    v8f   __attribute__((ext_vector_type(8)));

__device__ __forceinline__ v8f zero8() {
  v8f z;
#pragma unroll
  for (int i = 0; i < 8; ++i) z[i] = 0.f;
  return z;
}

// A fragment (16x32 f16, MxK): lane&15 = row (row ptr pre-resolved per lane),
// lane>>4 selects K sub-blocks {0..15} vs {16..31} (contiguous, matches B).
__device__ __forceinline__ v16h frag_a_f16(const _Float16* rowp, int k0, int lane) {
  int kb = k0 + ((lane >> 4) << 4);
  const _Float16* p = rowp + kb;
  half8 lo = *(const half8*)(p);
  half8 hi = *(const half8*)(p + 8);
  v16h a;
#pragma unroll
  for (int i = 0; i < 8; ++i) { a[i] = lo[i]; a[8 + i] = hi[i]; }
  return a;
}

// B fragment (32x16 f16, KxN) from TRANSPOSED weight T[col][K] (ldk = K):
// lane&15 = column, 16 contiguous K halves per lane (lanes>=16 get K+16..31).
__device__ __forceinline__ v16h frag_b_T(const _Float16* T, int ldk, int n0, int k0, int lane) {
  const _Float16* p = T + (n0 + (lane & 15)) * ldk + k0 + ((lane >> 4) << 4);
  half8 lo = *(const half8*)(p);
  half8 hi = *(const half8*)(p + 8);
  v16h b;
#pragma unroll
  for (int i = 0; i < 8; ++i) { b[i] = lo[i]; b[8 + i] = hi[i]; }
  return b;
}

#define WMMA_F16(a, b, c) \
  __builtin_amdgcn_wmma_f32_16x16x32_f16(false, (a), false, (b), (short)0, (c), false, false)

// ---------------------------------------------------------------------------
// Weight convert: W (K x Nc, row-major f32) -> T[c*K + k] f16 (transposed).
__global__ void wcvt_kernel(const float* __restrict__ W, _Float16* __restrict__ T,
                            int K, int Nc) {
  int idx = blockIdx.x * 256 + threadIdx.x;
  if (idx >= K * Nc) return;
  int k = idx / Nc, c = idx - k * Nc;
  T[c * K + k] = (_Float16)W[idx];
}

// Elementwise f32 -> f16 (activation pre-convert so GEMM A-side is pure b128).
__global__ void cvt_h_kernel(const float* __restrict__ src, _Float16* __restrict__ dst,
                             int n) {
  int idx = blockIdx.x * 256 + threadIdx.x;
  if (idx < n) dst[idx] = (_Float16)src[idx];
}

// ---------------------------------------------------------------------------
// node = relu(x @ W_enc + b_enc).  2048x512x256, WMMA f16, f32 accum.
// One wave computes 16 rows x 64 cols (4 accumulators): A loaded once per
// K-step, reused by 4 WMMAs.
__global__ __launch_bounds__(256) void encode_kernel(
    const _Float16* __restrict__ xh, const _Float16* __restrict__ WencT,
    const float* __restrict__ b_enc, float* __restrict__ node_f,
    _Float16* __restrict__ node_h) {
  int lane = threadIdx.x & 31;
  int gid  = blockIdx.x * 8 + (threadIdx.x >> 5);  // 512 wave-tiles: 128 rt x 4 cg
  int m0 = (gid >> 2) << 4;
  int cg = (gid & 3) << 6;  // column group base (64 cols)
  const _Float16* arow = xh + (m0 + (lane & 15)) * D_IN;
  v8f acc[4];
#pragma unroll
  for (int t = 0; t < 4; ++t) acc[t] = zero8();
#pragma unroll
  for (int kt = 0; kt < D_IN / 32; ++kt) {
    int k0 = kt * 32;
    v16h a = frag_a_f16(arow, k0, lane);
#pragma unroll
    for (int t = 0; t < 4; ++t)
      acc[t] = WMMA_F16(a, frag_b_T(WencT, D_IN, cg + t * 16, k0, lane), acc[t]);
  }
  int rb = m0 + ((lane >> 4) << 3);
#pragma unroll
  for (int t = 0; t < 4; ++t) {
    int col = cg + t * 16 + (lane & 15);
    float bias = b_enc[col];
#pragma unroll
    for (int v = 0; v < 8; ++v) {
      float val = fmaxf(acc[t][v] + bias, 0.f);
      node_f[(rb + v) * D_EMB + col] = val;
      node_h[(rb + v) * D_EMB + col] = (_Float16)val;
    }
  }
}

// ---------------------------------------------------------------------------
// a1[e] = relu([node[src]|node[dst]] @ W1a + b1a) @ W2a + b2a.
// Per wave: 16 edges x 64 hidden (4 accum tiles), K=512 as two K=256 passes
// (src half then dst half) sharing accumulators. W1aT K-half staged in LDS
// via CDNA5 async global->LDS DMA (ASYNCcnt).
__global__ __launch_bounds__(256) void edge_hidden_kernel(
    const _Float16* __restrict__ node_h, const int* __restrict__ ei,
    const _Float16* __restrict__ W1aT, const float* __restrict__ b1a,
    const float* __restrict__ W2a, const float* __restrict__ b2a,
    float* __restrict__ a1) {
  __shared__ _Float16 sW[64 * 256];  // 32KB: one K-half of W1aT, [col][k]
  int tid = threadIdx.x, lane = tid & 31, wave = tid >> 5;
  int e0 = blockIdx.x * 128 + wave * 16;
  unsigned lds_base = (unsigned)(size_t)(&sW[0]);  // LDS offset = addr[31:0]
  v8f acc[4];
#pragma unroll
  for (int t = 0; t < 4; ++t) acc[t] = zero8();

  for (int half = 0; half < 2; ++half) {
    __syncthreads();  // all waves done with previous sW contents
    // Async DMA: 2048 x 16B chunks, per-lane LDS dest + 32-bit global offset.
    for (int i = tid; i < 64 * 32; i += 256) {  // 64 cols * 32 chunks of 8 halves
      int c = i >> 5, kk = (i & 31) << 3;
      unsigned ldsaddr = lds_base + (unsigned)((c * 256 + kk) * 2);
      unsigned goff    = (unsigned)((c * 512 + half * 256 + kk) * 2);
      asm volatile("global_load_async_to_lds_b128 %0, %1, %2"
                   :: "v"(ldsaddr), "v"(goff), "s"(W1aT)
                   : "memory");
    }
    asm volatile("s_wait_asynccnt 0" ::: "memory");
    __syncthreads();

    int row = ei[half * NE + e0 + (lane & 15)];  // src (half 0) / dst (half 1)
    const _Float16* arow = node_h + row * D_EMB;
#pragma unroll
    for (int kt = 0; kt < 8; ++kt) {
      int k0 = kt * 32;
      v16h a = frag_a_f16(arow, k0, lane);
#pragma unroll
      for (int t = 0; t < 4; ++t) {
        const _Float16* p = sW + (t * 16 + (lane & 15)) * 256 + k0 + ((lane >> 4) << 4);
        half8 lo = *(const half8*)(p);
        half8 hi = *(const half8*)(p + 8);
        v16h b;
#pragma unroll
        for (int i = 0; i < 8; ++i) { b[i] = lo[i]; b[8 + i] = hi[i]; }
        acc[t] = WMMA_F16(a, b, acc[t]);
      }
    }
  }
  // Fold 64->1 projection: relu(h + b1a) * W2a, reduce across the 16 lanes
  // holding the hidden channels of each edge.
  float part[8];
#pragma unroll
  for (int v = 0; v < 8; ++v) part[v] = 0.f;
#pragma unroll
  for (int t = 0; t < 4; ++t) {
    int ch = t * 16 + (lane & 15);
    float w2 = W2a[ch], bb = b1a[ch];
#pragma unroll
    for (int v = 0; v < 8; ++v) part[v] += fmaxf(acc[t][v] + bb, 0.f) * w2;
  }
#pragma unroll
  for (int off = 1; off < 16; off <<= 1) {
#pragma unroll
    for (int v = 0; v < 8; ++v) part[v] += __shfl_xor(part[v], off, 32);
  }
  if ((lane & 15) == 0) {
    int eb = e0 + ((lane >> 4) << 3);
    float b2 = b2a[0];
#pragma unroll
    for (int v = 0; v < 8; ++v) a1[eb + v] = part[v] + b2;
  }
}

// ---------------------------------------------------------------------------
// Tiny per-edge MLPs: a2 = geo(8->16->1), edge_emb = fuse([a1,a2]: 2->8->1).
__global__ void edge_small_kernel(
    const float* __restrict__ coords, const int* __restrict__ ei,
    const float* __restrict__ a1,
    const float* W1g, const float* b1g, const float* W2g, const float* b2g,
    const float* W1f, const float* b1f, const float* W2f, const float* b2f,
    float* __restrict__ emb) {
  int e = blockIdx.x * 256 + threadIdx.x;
  if (e >= NE) return;
  int s = ei[e], d = ei[NE + e];
  float g[8];
#pragma unroll
  for (int i = 0; i < 4; ++i) { g[i] = coords[s * 4 + i]; g[4 + i] = coords[d * 4 + i]; }
  float a2 = b2g[0];
#pragma unroll
  for (int j = 0; j < 16; ++j) {
    float h = b1g[j];
#pragma unroll
    for (int i = 0; i < 8; ++i) h += g[i] * W1g[i * 16 + j];
    a2 += fmaxf(h, 0.f) * W2g[j];
  }
  float v1 = a1[e];
  float em = b2f[0];
#pragma unroll
  for (int j = 0; j < 8; ++j) {
    float h = b1f[j] + v1 * W1f[j] + a2 * W1f[8 + j];
    em += fmaxf(h, 0.f) * W2f[j];
  }
  emb[e] = em;
}

// ---------------------------------------------------------------------------
// msg = segment_sum(edge_emb * node[src], dst). Scatter-add with fp32 atomics.
__global__ void message_kernel(const float* __restrict__ node_f,
                               const int* __restrict__ ei,
                               const float* __restrict__ emb,
                               float* __restrict__ msg) {
  int idx = blockIdx.x * 256 + threadIdx.x;  // NE*64 work-items, 4 dims each
  if (idx >= NE * 64) return;
  int e = idx >> 6;
  int d = (idx & 63) << 2;
  int s = ei[e], t = ei[NE + e];
  float w = emb[e];
  float4 nv = *(const float4*)(node_f + s * D_EMB + d);
  float* mp = msg + t * D_EMB + d;
  __hip_atomic_fetch_add(mp + 0, w * nv.x, __ATOMIC_RELAXED, __HIP_MEMORY_SCOPE_AGENT);
  __hip_atomic_fetch_add(mp + 1, w * nv.y, __ATOMIC_RELAXED, __HIP_MEMORY_SCOPE_AGENT);
  __hip_atomic_fetch_add(mp + 2, w * nv.z, __ATOMIC_RELAXED, __HIP_MEMORY_SCOPE_AGENT);
  __hip_atomic_fetch_add(mp + 3, w * nv.w, __ATOMIC_RELAXED, __HIP_MEMORY_SCOPE_AGENT);
}

// ---------------------------------------------------------------------------
// out = relu(node @ Wn + msg @ Wm + bo). Fused K=512 WMMA loop, 16x64 per wave.
__global__ __launch_bounds__(256) void out_kernel(
    const _Float16* __restrict__ node_h, const _Float16* __restrict__ msg_h,
    const _Float16* __restrict__ WnT, const _Float16* __restrict__ WmT,
    const float* __restrict__ bo, float* __restrict__ outf) {
  int lane = threadIdx.x & 31;
  int gid  = blockIdx.x * 8 + (threadIdx.x >> 5);  // 512 wave-tiles
  int m0 = (gid >> 2) << 4;
  int cg = (gid & 3) << 6;
  int ar = m0 + (lane & 15);
  const _Float16* ah = node_h + ar * D_EMB;
  const _Float16* am = msg_h  + ar * D_EMB;
  v8f acc[4];
#pragma unroll
  for (int t = 0; t < 4; ++t) acc[t] = zero8();
#pragma unroll
  for (int kt = 0; kt < 8; ++kt) {
    int k0 = kt * 32;
    v16h a = frag_a_f16(ah, k0, lane);
#pragma unroll
    for (int t = 0; t < 4; ++t)
      acc[t] = WMMA_F16(a, frag_b_T(WnT, D_EMB, cg + t * 16, k0, lane), acc[t]);
  }
#pragma unroll
  for (int kt = 0; kt < 8; ++kt) {
    int k0 = kt * 32;
    v16h a = frag_a_f16(am, k0, lane);
#pragma unroll
    for (int t = 0; t < 4; ++t)
      acc[t] = WMMA_F16(a, frag_b_T(WmT, D_EMB, cg + t * 16, k0, lane), acc[t]);
  }
  int rb = m0 + ((lane >> 4) << 3);
#pragma unroll
  for (int t = 0; t < 4; ++t) {
    int col = cg + t * 16 + (lane & 15);
    float bias = bo[col];
#pragma unroll
    for (int v = 0; v < 8; ++v)
      outf[(rb + v) * D_EMB + col] = fmaxf(acc[t][v] + bias, 0.f);
  }
}

// ---------------------------------------------------------------------------
// M init: slack = exp(SLACK*LAM) = exp(-1) on last row/col, 0 elsewhere.
__global__ void init_M_kernel(float* __restrict__ M) {
  int idx = blockIdx.x * 256 + threadIdx.x;
  if (idx >= MROWS * MCOLS) return;
  int r = idx / MCOLS, c = idx - r * MCOLS;
  M[idx] = (r < T_NUM && c < DET_NUM) ? 0.f : __expf(-1.0f);
}

// One wave per edge: cosine sim (256-dim), IoU, affinity MLP, exp -> M scatter.
__global__ __launch_bounds__(256) void affinity_kernel(
    const float* __restrict__ outf, const float* __restrict__ cbox,
    const int* __restrict__ ei,
    const float* W1c, const float* b1c, const float* W2c, const float* b2c,
    float* __restrict__ M) {
  int lane = threadIdx.x & 31;
  int e = blockIdx.x * 8 + (threadIdx.x >> 5);  // e in [0, E_EDGES)
  int sf = ei[e], df = ei[NE + e];
  const float* oa = outf + sf * D_EMB;
  const float* ob = outf + df * D_EMB;
  float dot = 0.f, na = 0.f, nb = 0.f;
#pragma unroll
  for (int i = lane; i < D_EMB; i += 32) {
    float a = oa[i], b = ob[i];
    dot += a * b; na += a * a; nb += b * b;
  }
#pragma unroll
  for (int off = 16; off > 0; off >>= 1) {
    dot += __shfl_xor(dot, off, 32);
    na  += __shfl_xor(na,  off, 32);
    nb  += __shfl_xor(nb,  off, 32);
  }
  if (lane == 0) {
    float denom = fmaxf(sqrtf(na) * sqrtf(nb), 1e-6f);
    float cosv = dot / denom;
    const float* A = cbox + sf * 4;
    const float* B = cbox + df * 4;
    float ltx = fmaxf(A[0], B[0]), lty = fmaxf(A[1], B[1]);
    float rbx = fminf(A[2], B[2]), rby = fminf(A[3], B[3]);
    float w = fmaxf(rbx - ltx, 0.f), h = fmaxf(rby - lty, 0.f);
    float inter = w * h;
    float areaA = (A[2] - A[0]) * (A[3] - A[1]);
    float areaB = (B[2] - B[0]) * (B[3] - B[1]);
    float iou = inter / (areaA + areaB - inter + 1e-9f);
    float aff = b2c[0];
#pragma unroll
    for (int j = 0; j < 8; ++j) {
      float hh = b1c[j] + cosv * W1c[j] + iou * W1c[8 + j];
      aff += fmaxf(hh, 0.f) * W2c[j];
    }
    M[sf * MCOLS + (df - T_NUM)] = __expf(aff * 5.0f);
  }
}

// ---------------------------------------------------------------------------
// Sinkhorn normalization passes.
__device__ __forceinline__ float block_sum_broadcast(float v) {
#pragma unroll
  for (int off = 16; off > 0; off >>= 1) v += __shfl_xor(v, off, 32);
  __shared__ float red[8];
  __shared__ float tot;
  int lane = threadIdx.x & 31, w = threadIdx.x >> 5;
  if (lane == 0) red[w] = v;
  __syncthreads();
  if (threadIdx.x == 0) {
    float t = 0.f;
    for (int i = 0; i < 8; ++i) t += red[i];
    tot = t;
  }
  __syncthreads();
  return tot;
}

__global__ __launch_bounds__(256) void sinkhorn_row(float* __restrict__ M) {
  int r = blockIdx.x;
  float s = 0.f;
  for (int c = threadIdx.x; c < MCOLS; c += 256) s += M[r * MCOLS + c];
  float inv = 1.0f / (block_sum_broadcast(s) + 1e-9f);
  for (int c = threadIdx.x; c < MCOLS; c += 256) M[r * MCOLS + c] *= inv;
}

__global__ __launch_bounds__(256) void sinkhorn_col(float* __restrict__ M) {
  int c = blockIdx.x;
  float s = 0.f;
  for (int r = threadIdx.x; r < MROWS; r += 256) s += M[r * MCOLS + c];
  float inv = 1.0f / (block_sum_broadcast(s) + 1e-9f);
  for (int r = threadIdx.x; r < MROWS; r += 256) M[r * MCOLS + c] *= inv;
}

// ---------------------------------------------------------------------------
// Pack output tuple: M[:1024,:1024] flat, ground_truth, det_num, tracklet_num.
__global__ void pack_kernel(const float* __restrict__ M, const float* __restrict__ gt,
                            float* __restrict__ outp) {
  int idx = blockIdx.x * 256 + threadIdx.x;
  if (idx < T_NUM * DET_NUM) {
    int r = idx >> 10, c = idx & 1023;
    outp[idx] = M[r * MCOLS + c];
    outp[T_NUM * DET_NUM + idx] = gt[idx];
  }
  if (idx == 0) {
    int* ip = (int*)(outp + 2 * T_NUM * DET_NUM);
    ip[0] = DET_NUM;
    ip[1] = T_NUM;
  }
}

// ---------------------------------------------------------------------------
extern "C" void kernel_launch(void* const* d_in, const int* in_sizes, int n_in,
                              void* d_out, int out_size, void* d_ws, size_t ws_size,
                              hipStream_t stream) {
  const float* x      = (const float*)d_in[0];
  const float* coords = (const float*)d_in[1];
  const float* cbox   = (const float*)d_in[2];
  const float* gt     = (const float*)d_in[3];
  const int*   ei     = (const int*)d_in[4];
  const float* W_enc = (const float*)d_in[5];  const float* b_enc = (const float*)d_in[6];
  const float* W1a   = (const float*)d_in[7];  const float* b1a   = (const float*)d_in[8];
  const float* W2a   = (const float*)d_in[9];  const float* b2a   = (const float*)d_in[10];
  const float* W1g   = (const float*)d_in[11]; const float* b1g   = (const float*)d_in[12];
  const float* W2g   = (const float*)d_in[13]; const float* b2g   = (const float*)d_in[14];
  const float* W1f   = (const float*)d_in[15]; const float* b1f   = (const float*)d_in[16];
  const float* W2f   = (const float*)d_in[17]; const float* b2f   = (const float*)d_in[18];
  const float* Wn    = (const float*)d_in[19]; const float* Wm    = (const float*)d_in[20];
  const float* bo    = (const float*)d_in[21];
  const float* W1c   = (const float*)d_in[22]; const float* b1c   = (const float*)d_in[23];
  const float* W2c   = (const float*)d_in[24]; const float* b2c   = (const float*)d_in[25];
  (void)in_sizes; (void)n_in; (void)out_size; (void)ws_size;

  char* ws = (char*)d_ws;
  size_t off = 0;
  auto take = [&](size_t bytes) -> char* {
    char* p = ws + off;
    off = (off + bytes + 255) & ~(size_t)255;
    return p;
  };
  _Float16* WencT  = (_Float16*)take((size_t)D_IN * D_EMB * 2);
  _Float16* W1aT   = (_Float16*)take((size_t)D_IN * 64 * 2);
  _Float16* WnT    = (_Float16*)take((size_t)D_EMB * D_EMB * 2);
  _Float16* WmT    = (_Float16*)take((size_t)D_EMB * D_EMB * 2);
  _Float16* xh     = (_Float16*)take((size_t)NN * D_IN * 2);
  float*    node_f = (float*)take((size_t)NN * D_EMB * 4);
  _Float16* node_h = (_Float16*)take((size_t)NN * D_EMB * 2);
  float*    msg    = (float*)take((size_t)NN * D_EMB * 4);
  _Float16* msg_h  = (_Float16*)take((size_t)NN * D_EMB * 2);
  float*    outf   = (float*)take((size_t)NN * D_EMB * 4);
  float*    a1     = (float*)take((size_t)NE * 4);
  float*    emb    = (float*)take((size_t)NE * 4);
  float*    M      = (float*)take((size_t)MROWS * MCOLS * 4);

  // 1) One-time weight convert+transpose to f16; activation x -> f16.
  wcvt_kernel<<<(D_IN * D_EMB + 255) / 256, 256, 0, stream>>>(W_enc, WencT, D_IN, D_EMB);
  wcvt_kernel<<<(D_IN * 64 + 255) / 256, 256, 0, stream>>>(W1a, W1aT, D_IN, 64);
  wcvt_kernel<<<(D_EMB * D_EMB + 255) / 256, 256, 0, stream>>>(Wn, WnT, D_EMB, D_EMB);
  wcvt_kernel<<<(D_EMB * D_EMB + 255) / 256, 256, 0, stream>>>(Wm, WmT, D_EMB, D_EMB);
  cvt_h_kernel<<<(NN * D_IN + 255) / 256, 256, 0, stream>>>(x, xh, NN * D_IN);

  // 2) Node encoder GEMM (WMMA, 16x64 per wave).
  encode_kernel<<<64, 256, 0, stream>>>(xh, WencT, b_enc, node_f, node_h);

  // 3) Edge attention MLP (dominant GEMM, WMMA + async-LDS staged weights).
  edge_hidden_kernel<<<NE / 128, 256, 0, stream>>>(node_h, ei, W1aT, b1a, W2a, b2a, a1);

  // 4) Tiny geo/fuse MLPs per edge.
  edge_small_kernel<<<(NE + 255) / 256, 256, 0, stream>>>(
      coords, ei, a1, W1g, b1g, W2g, b2g, W1f, b1f, W2f, b2f, emb);

  // 5) Message passing: scatter-add, then f16 convert for the next GEMM.
  hipMemsetAsync(msg, 0, (size_t)NN * D_EMB * 4, stream);
  message_kernel<<<(NE * 64) / 256, 256, 0, stream>>>(node_f, ei, emb, msg);
  cvt_h_kernel<<<(NN * D_EMB + 255) / 256, 256, 0, stream>>>(msg, msg_h, NN * D_EMB);

  // 6) Output projection: relu(node@Wn + msg@Wm + bo) (WMMA, fused K loop).
  out_kernel<<<64, 256, 0, stream>>>(node_h, msg_h, WnT, WmT, bo, outf);

  // 7) Cost matrix + Sinkhorn.
  init_M_kernel<<<(MROWS * MCOLS + 255) / 256, 256, 0, stream>>>(M);
  affinity_kernel<<<E_EDGES / 8, 256, 0, stream>>>(outf, cbox, ei, W1c, b1c, W2c, b2c, M);
  for (int it = 0; it < 8; ++it) {
    sinkhorn_row<<<MROWS, 256, 0, stream>>>(M);
    sinkhorn_col<<<MCOLS, 256, 0, stream>>>(M);
  }

  // 8) Pack tuple output.
  pack_kernel<<<(T_NUM * DET_NUM + 255) / 256, 256, 0, stream>>>(M, gt, (float*)d_out);
}
